// CrossAttentionGAT_30648886624773
// MI455X (gfx1250) — compile-verified, matrix operand order checked by reading, not run on hardware
//
#include <hip/hip_runtime.h>

typedef __attribute__((ext_vector_type(16))) _Float16 v16h;
typedef __attribute__((ext_vector_type(8)))  float    v8f;

#define HC 1024   // H*C
#define HH 8      // heads
#define CC 128    // channels per head
#define KIN 128   // input feature dim
#define NEG_SLOPE 0.2f

// ---------- helpers: order-preserving float <-> uint for atomic max ----------
__device__ __forceinline__ unsigned f2ord(float f) {
  unsigned u = __float_as_uint(f);
  return (u & 0x80000000u) ? ~u : (u | 0x80000000u);
}
__device__ __forceinline__ float ord2f(unsigned u) {
  u = (u & 0x80000000u) ? (u & 0x7FFFFFFFu) : ~u;
  return __uint_as_float(u);
}

// ---------- conversion kernels ----------
__global__ void k_f32_to_f16(const float* __restrict__ in, _Float16* __restrict__ out, int n) {
  int i = blockIdx.x * blockDim.x + threadIdx.x;
  if (i < n) out[i] = (_Float16)in[i];
}

// Wt[n*KIN + k] = W[k*HC + n], as f16  (W is [KIN, HC] row-major)
__global__ void k_transpose_f16(const float* __restrict__ W, _Float16* __restrict__ Wt, int total) {
  int i = blockIdx.x * blockDim.x + threadIdx.x;
  if (i < total) {
    int n = i / KIN, k = i - n * KIN;
    Wt[i] = (_Float16)W[(size_t)k * HC + n];
  }
}

// ---------- WMMA GEMM: C[M,HC] = A[M,KIN] (f16) x Bt[HC,KIN]^T (f16), f32 acc ----------
// One wave computes one 16x16 tile; K=128 -> 4 x v_wmma_f32_16x16x32_f16.
// 16-bit A fragment per-lane layout (ISA 7.12.2): lane L holds row M=L%16, half=L/16;
// dword p (p=0..7) of the v16h holds K pair at K = (p/4)*16 + half*8 + (p%4)*2.
// B fragment mirrors with column N = L%16 (W pre-transposed so pairs are contiguous).
__global__ __launch_bounds__(256)
void k_gemm_wmma(const _Float16* __restrict__ A, const _Float16* __restrict__ Bt,
                 float* __restrict__ C, int M) {
  const int lane = threadIdx.x & 31;
  const int wave = threadIdx.x >> 5;
  const int gw   = blockIdx.x * (blockDim.x >> 5) + wave;
  const int ntiles = HC / 16;                 // 64
  const int mt = gw / ntiles, nt = gw % ntiles;
  if (mt * 16 >= M) return;
  const int half = lane >> 4;
  const int l16  = lane & 15;

  const unsigned* arow = (const unsigned*)(A  + (size_t)(mt * 16 + l16) * KIN);
  const unsigned* brow = (const unsigned*)(Bt + (size_t)(nt * 16 + l16) * KIN);

  v8f c = {};
  #pragma unroll
  for (int ks = 0; ks < 4; ++ks) {            // K step of 32
    union { v16h v; unsigned u[8]; } a, b;
    const int base = ks * 16 + half * 4;      // dword base within the 64-dword row
    #pragma unroll
    for (int p = 0; p < 8; ++p) {
      const int dw = base + (p >> 2) * 8 + (p & 3);
      a.u[p] = arow[dw];
      b.u[p] = brow[dw];
    }
    c = __builtin_amdgcn_wmma_f32_16x16x32_f16(false, a.v, false, b.v,
                                               (short)0, c, false, false);
  }
  float* crow = C + (size_t)(mt * 16) * HC + nt * 16 + l16;
  #pragma unroll
  for (int r = 0; r < 8; ++r)
    crow[(size_t)(r + 8 * half) * HC] = c[r];
}

// ---------- attention coefficients: one wave per (node, head) ----------
__global__ __launch_bounds__(256)
void k_attn_coef(const float* __restrict__ h, const float* __restrict__ att_src,
                 const float* __restrict__ att_dst, float* __restrict__ a_src,
                 float* __restrict__ a_dst, int N) {
  const int lane = threadIdx.x & 31;
  const int gw = blockIdx.x * (blockDim.x >> 5) + (threadIdx.x >> 5);
  const int n = gw >> 3, head = gw & 7;
  if (n >= N) return;
  const float* hp = h + (size_t)n * HC + head * CC;
  const float* as = att_src + head * CC;
  const float* ad = att_dst + head * CC;
  float ss = 0.f, sd = 0.f;
  #pragma unroll
  for (int c = lane; c < CC; c += 32) { float v = hp[c]; ss += v * as[c]; sd += v * ad[c]; }
  #pragma unroll
  for (int off = 16; off; off >>= 1) {
    ss += __shfl_down(ss, off, 32);
    sd += __shfl_down(sd, off, 32);
  }
  if (lane == 0) { a_src[n * HH + head] = ss; a_dst[n * HH + head] = sd; }
}

// ---------- init ----------
__global__ void k_init_nodes(unsigned* __restrict__ mkey, float* __restrict__ den,
                             float* __restrict__ s, int n8) {
  int i = blockIdx.x * blockDim.x + threadIdx.x;
  if (i < n8) { mkey[i] = 0x007FFFFFu /* f2ord(-inf) */; den[i] = 0.f; s[i] = 0.f; }
}
__global__ void k_zero(float* __restrict__ p, int n) {
  int i = blockIdx.x * blockDim.x + threadIdx.x;
  if (i < n) p[i] = 0.f;
}

// ---------- edge passes (items: t<E real edges, t>=E self loops) ----------
__device__ __forceinline__ void edge_sd(const long long* ei, int E, int t, int& s, int& d) {
  if (t < E) { s = (int)ei[t]; d = (int)ei[E + t]; } else { s = d = t - E; }
}

__global__ void k_pass_max(const long long* __restrict__ ei, int E, int N,
                           const float* __restrict__ asrc, const float* __restrict__ adst,
                           unsigned* __restrict__ mkey) {
  int t = blockIdx.x * blockDim.x + threadIdx.x;
  if (t >= E + N) return;
  int s, d; edge_sd(ei, E, t, s, d);
  #pragma unroll
  for (int h = 0; h < HH; ++h) {
    float v = asrc[s * HH + h] + adst[d * HH + h];
    v = v > 0.f ? v : NEG_SLOPE * v;
    atomicMax(&mkey[d * HH + h], f2ord(v));
  }
}

__global__ void k_pass_den(const long long* __restrict__ ei, int E, int N,
                           const float* __restrict__ asrc, const float* __restrict__ adst,
                           const unsigned* __restrict__ mkey, float* __restrict__ den) {
  int t = blockIdx.x * blockDim.x + threadIdx.x;
  if (t >= E + N) return;
  int s, d; edge_sd(ei, E, t, s, d);
  #pragma unroll
  for (int h = 0; h < HH; ++h) {
    float v = asrc[s * HH + h] + adst[d * HH + h];
    v = v > 0.f ? v : NEG_SLOPE * v;
    atomicAdd(&den[d * HH + h], expf(v - ord2f(mkey[d * HH + h])));
  }
}

__global__ void k_pass_alpha(const long long* __restrict__ ei, int E, int N,
                             const float* __restrict__ asrc, const float* __restrict__ adst,
                             const unsigned* __restrict__ mkey, const float* __restrict__ den,
                             float* __restrict__ sw) {
  int t = blockIdx.x * blockDim.x + threadIdx.x;
  if (t >= E + N) return;
  int s, d; edge_sd(ei, E, t, s, d);
  #pragma unroll
  for (int h = 0; h < HH; ++h) {
    float v = asrc[s * HH + h] + adst[d * HH + h];
    v = v > 0.f ? v : NEG_SLOPE * v;
    float alpha = expf(v - ord2f(mkey[d * HH + h])) / den[d * HH + h];
    atomicAdd(&sw[s * HH + h], alpha);   // accumulate per-source alpha mass
  }
}

// ---------- weighted column reduction: acc[c] += sum_n sw[n, c/128] * h[n, c] ----------
__global__ __launch_bounds__(1024)
void k_colsum(const float* __restrict__ h, const float* __restrict__ sw,
              float* __restrict__ acc, int N, int rowsPerBlock) {
  const int c = threadIdx.x;          // 0..1023
  const int head = c >> 7;
  int r0 = blockIdx.x * rowsPerBlock;
  int r1 = r0 + rowsPerBlock; if (r1 > N) r1 = N;
  float sum = 0.f;
  for (int n = r0; n < r1; ++n) sum += sw[n * HH + head] * h[(size_t)n * HC + c];
  atomicAdd(&acc[c], sum);
}

// ---------- final projection: out[j] = (acc/N + bias) . Wl[:,j] + bl[j] ----------
__global__ __launch_bounds__(128)
void k_final(const float* __restrict__ acc, const float* __restrict__ bias,
             const float* __restrict__ Wl, const float* __restrict__ bl,
             float* __restrict__ out, float invN) {
  const int j = threadIdx.x;          // 0..127
  float sum = 0.f;
  for (int c = 0; c < HC; ++c) {
    float m = acc[c] * invN + bias[c];
    sum += m * Wl[(size_t)c * 128 + j];
  }
  out[j] = sum + bl[j];
}

// ---------- host-side pipeline for one graph ----------
static void run_graph(const float* x, const long long* ei, const float* W,
                      const float* att_src, const float* att_dst,
                      float* h, _Float16* xh, _Float16* wt,
                      float* a_src, float* a_dst, unsigned* mkey, float* den,
                      float* sw, float* acc, int N, int E, hipStream_t stream) {
  const int B = 256;
  // convert inputs to f16 (W transposed for B-fragment contiguity)
  k_f32_to_f16<<<(N * KIN + B - 1) / B, B, 0, stream>>>(x, xh, N * KIN);
  k_transpose_f16<<<(HC * KIN + B - 1) / B, B, 0, stream>>>(W, wt, HC * KIN);
  // h = x @ W via WMMA
  int totalWaves = (N / 16) * (HC / 16);
  k_gemm_wmma<<<(totalWaves + 7) / 8, 256, 0, stream>>>(xh, wt, h, N);
  // per-node attention logits
  int coefWaves = N * HH;
  k_attn_coef<<<(coefWaves + 7) / 8, 256, 0, stream>>>(h, att_src, att_dst, a_src, a_dst, N);
  // edge softmax (3 passes over E real edges + N self loops)
  k_init_nodes<<<(N * HH + B - 1) / B, B, 0, stream>>>(mkey, den, sw, N * HH);
  int items = E + N;
  k_pass_max  <<<(items + B - 1) / B, B, 0, stream>>>(ei, E, N, a_src, a_dst, mkey);
  k_pass_den  <<<(items + B - 1) / B, B, 0, stream>>>(ei, E, N, a_src, a_dst, mkey, den);
  k_pass_alpha<<<(items + B - 1) / B, B, 0, stream>>>(ei, E, N, a_src, a_dst, mkey, den, sw);
  // mean(emb)*N  (bias added in k_final)
  k_zero<<<(HC + B - 1) / B, B, 0, stream>>>(acc, HC);
  const int rowsPerBlock = 128;
  k_colsum<<<(N + rowsPerBlock - 1) / rowsPerBlock, 1024, 0, stream>>>(h, sw, acc, N, rowsPerBlock);
}

extern "C" void kernel_launch(void* const* d_in, const int* in_sizes, int n_in,
                              void* d_out, int out_size, void* d_ws, size_t ws_size,
                              hipStream_t stream) {
  const float*     x1   = (const float*)d_in[0];
  const long long* ei1  = (const long long*)d_in[1];
  const float*     W1   = (const float*)d_in[2];
  const float*     as1  = (const float*)d_in[3];
  const float*     ad1  = (const float*)d_in[4];
  const float*     b1   = (const float*)d_in[5];
  const float*     x2   = (const float*)d_in[6];
  const long long* ei2  = (const long long*)d_in[7];
  const float*     W2   = (const float*)d_in[8];
  const float*     as2  = (const float*)d_in[9];
  const float*     ad2  = (const float*)d_in[10];
  const float*     b2   = (const float*)d_in[11];
  const float*     Wl   = (const float*)d_in[12];
  const float*     bl   = (const float*)d_in[13];
  float*           out  = (float*)d_out;

  const int N = in_sizes[0] / KIN;   // 8192
  const int E = in_sizes[1] / 2;     // 262144

  // carve workspace (reused sequentially for graph 1 then graph 2)
  size_t off = 0;
  auto carve = [&](size_t bytes) {
    void* p = (char*)d_ws + off;
    off += (bytes + 255) & ~(size_t)255;
    return p;
  };
  float*    h     = (float*)   carve((size_t)N * HC * sizeof(float));
  _Float16* xh    = (_Float16*)carve((size_t)N * KIN * sizeof(_Float16));
  _Float16* wt    = (_Float16*)carve((size_t)HC * KIN * sizeof(_Float16));
  float*    a_src = (float*)   carve((size_t)N * HH * sizeof(float));
  float*    a_dst = (float*)   carve((size_t)N * HH * sizeof(float));
  unsigned* mkey  = (unsigned*)carve((size_t)N * HH * sizeof(unsigned));
  float*    den   = (float*)   carve((size_t)N * HH * sizeof(float));
  float*    sw    = (float*)   carve((size_t)N * HH * sizeof(float));
  float*    acc1  = (float*)   carve(HC * sizeof(float));
  float*    acc2  = (float*)   carve(HC * sizeof(float));
  (void)ws_size; (void)n_in; (void)out_size;

  run_graph(x1, ei1, W1, as1, ad1, h, xh, wt, a_src, a_dst, mkey, den, sw, acc1, N, E, stream);
  run_graph(x2, ei2, W2, as2, ad2, h, xh, wt, a_src, a_dst, mkey, den, sw, acc2, N, E, stream);

  const float invN = 1.0f / (float)N;
  // mean(cross1) uses emb2; mean(cross2) uses emb1 (softmax columns/rows sum to 1)
  k_final<<<1, 128, 0, stream>>>(acc2, b2, Wl, bl, out, invN);
  k_final<<<1, 128, 0, stream>>>(acc1, b1, Wl, bl, out + 128, invN);
}
